// WindowedGraphAttention_44744969289763
// MI455X (gfx1250) — compile-verified, matrix operand order checked by reading, not run on hardware
//
#include <hip/hip_runtime.h>
#include <hip/hip_bf16.h>
#include <stdint.h>

// ---------------------------------------------------------------------------
// Fused windowed cross-attention for MI455X (gfx1250, wave32, WMMA + TDM).
//   out = x1 + gamma * ( (softmax(q k^T / sqrt(16)) v) * wp + bp )
// B=4, C=128, H=W=256, window 8x8 (64 tokens), Cq=16.
// One window per 128-thread block (4 wave32s); BF16 WMMA 16x16x32, f32 acc.
// x1/x2 window tiles are fetched by the Tensor Data Mover (tensor_load_to_lds)
// in channel-major [c][token] layout with a 1-dword pad every 64 dwords
// (stride 65 f32) for conflict-free epilogue reads.
// ---------------------------------------------------------------------------

typedef __attribute__((ext_vector_type(16))) __bf16    v16bf;
typedef __attribute__((ext_vector_type(8)))  float     v8f;
typedef __attribute__((ext_vector_type(4)))  uint32_t  v4u;
typedef __attribute__((ext_vector_type(8)))  uint32_t  v8u;

union ABf { v16bf v; uint32_t u[8]; };
union Acc { v8f v; float f[8]; };

// ---- LDS layout (dynamic shared, byte offsets; all 16B aligned) -----------
#define SXC_STRIDE  65                       // fp32 tile row pitch per channel
#define OFF_SX1     0                        // 128 x 65 f32   = 33280 (TDM dest)
#define OFF_SWQ     33280                    // 16 x 128 bf16  =  4096
#define OFF_SWK     37376                    // 16 x 128 bf16  =  4096
#define OFF_SWV     41472                    // 128 x 128 bf16 = 32768
#define OFF_SVT     74240                    // 128 x 72 bf16  = 18432 (v^T, padded)
#define OFF_SP      92672                    // 4 x (16 x 64) bf16 = 8192
#define OFF_SX2     100864                   // 128 x 65 f32   = 33280 (TDM dest)
#define OFF_SK      134144                   // 64 x 16 bf16   =  2048
#define OFF_SQ      136192                   // 4 x (16 x 16) bf16 = 2048
#define SMEM_BYTES  138240                   // 135 KB
#define OFF_SO      OFF_SX2                  // 128 x 65 f32, aliases dead sX2
#define SVT_STRIDE  72
#define SO_STRIDE   65

__device__ __forceinline__ uint32_t f2bf1(float f) {
  uint32_t u = __float_as_uint(f);
  return (u + 0x7FFFu + ((u >> 16) & 1u)) >> 16;   // RNE truncate to bf16
}
__device__ __forceinline__ uint32_t pack_bf(float lo, float hi) {
  return f2bf1(lo) | (f2bf1(hi) << 16);
}

// ---------------------------------------------------------------------------
// Tensor Data Mover: DMA one 8(w) x 8(h) x 128(c) fp32 tile, global -> LDS.
// LDS receives it packed in loop order (w,h,c) => channel-major [c][64 tokens],
// with pad_enable adding 1 dword after every 64 dwords => row pitch 65 f32.
// Descriptor per CDNA5 ISA ch.8; all fields block-uniform => SGPR groups.
// ---------------------------------------------------------------------------
__device__ __forceinline__ void tdm_load_tile(uint32_t lds_addr, const float* gptr) {
  const uint64_t ga = (uint64_t)(uintptr_t)gptr;
  v4u g0; v8u g1; v4u g2; v4u g3;
  // group0: count=1 (valid, load, user), lds_addr, 57b global addr, type=2
  g0[0] = 1u;
  g0[1] = lds_addr;
  g0[2] = (uint32_t)ga;
  g0[3] = (uint32_t)((ga >> 32) & 0x01FFFFFFu) | 0x80000000u;
  // group1: data_size=4B(code2), pad_enable, pad_interval=64dw(code5),
  //         pad_amount=1dw(code0); tile 8 x 8 x 128; big tensor dims (no OOB);
  //         dim0_stride=W=256, dim1_stride=H*W=65536
  g1[0] = (2u << 16) | (1u << 20) | (5u << 22);
  g1[1] = 0u;                      // atomic_barrier=0 | tensor_dim0 lo16 (2^30 -> 0)
  g1[2] = 0x4000u;                 // tensor_dim0 hi16 | tensor_dim1 lo16
  g1[3] = 0x4000u | (8u << 16);    // tensor_dim1 hi16 | tile_dim0 = 8 (w)
  g1[4] = 8u | (128u << 16);       // tile_dim1 = 8 (h) | tile_dim2 = 128 (c)
  g1[5] = 256u;                    // tensor_dim0_stride lo32 (= W)
  g1[6] = 0u;                      // dim0_stride hi16 | dim1_stride lo16
  g1[7] = 1u;                      // dim1_stride bits 47:16 (65536 >> 16)
  // group2: tensor_dim2 big; no 4th/5th dims
  g2[0] = 0x40000000u; g2[1] = 0u; g2[2] = 0u; g2[3] = 0u;
  g3[0] = 0u; g3[1] = 0u; g3[2] = 0u; g3[3] = 0u;
  asm volatile("tensor_load_to_lds %0, %1, %2, %3"
               :: "s"(g0), "s"(g1), "s"(g2), "s"(g3)
               : "memory");
}

// A/B fragment from bf16 LDS, row-major with `stride` (elements).
// ISA 16-bit layout: lane%16 = row (A:M, B:N); v0..3 hold K kb..kb+7,
// v4..7 hold K kb+16..kb+23, where kb = kbase + 8*(lane>=16).
__device__ __forceinline__ v16bf load_bf16_frag(const uint16_t* base, int stride,
                                                int lane, int kbase) {
  const int r  = lane & 15;
  const int kb = kbase + ((lane >> 4) << 3);
  const uint16_t* p = base + r * stride + kb;
  v4u lo = *(const v4u*)(p);
  v4u hi = *(const v4u*)(p + 16);
  ABf a;
  a.u[0] = lo.x; a.u[1] = lo.y; a.u[2] = lo.z; a.u[3] = lo.w;
  a.u[4] = hi.x; a.u[5] = hi.y; a.u[6] = hi.z; a.u[7] = hi.w;
  return a.v;
}

// Same, but logical K=16 zero-padded to 32 (for the q k^T matmul).
__device__ __forceinline__ v16bf load_bf16_frag16(const uint16_t* base, int stride,
                                                  int lane) {
  const int r  = lane & 15;
  const int kb = (lane >> 4) << 3;
  v4u lo = *(const v4u*)(base + r * stride + kb);
  ABf a;
  a.u[0] = lo.x; a.u[1] = lo.y; a.u[2] = lo.z; a.u[3] = lo.w;
  a.u[4] = 0; a.u[5] = 0; a.u[6] = 0; a.u[7] = 0;
  return a.v;
}

// A fragment from the channel-major fp32 TDM tile (gather stride SXC_STRIDE),
// with on-the-fly f32 -> bf16 conversion. lane%16 = token row within M-tile.
__device__ __forceinline__ v16bf load_a_cm(const float* base, int lane,
                                           int tok0, int kbase) {
  const int t  = tok0 + (lane & 15);
  const int kb = kbase + ((lane >> 4) << 3);
  const float* p = base + kb * SXC_STRIDE + t;
  ABf a;
#pragma unroll
  for (int i = 0; i < 4; ++i)
    a.u[i] = pack_bf(p[(2 * i) * SXC_STRIDE], p[(2 * i + 1) * SXC_STRIDE]);
  const float* p2 = p + 16 * SXC_STRIDE;
#pragma unroll
  for (int i = 0; i < 4; ++i)
    a.u[4 + i] = pack_bf(p2[(2 * i) * SXC_STRIDE], p2[(2 * i + 1) * SXC_STRIDE]);
  return a.v;
}

__device__ __forceinline__ v8f wmma_bf16(v16bf a, v16bf b, v8f c) {
  return __builtin_amdgcn_wmma_f32_16x16x32_bf16(false, a, false, b,
                                                 (short)0, c, false, false);
}

__global__ void __launch_bounds__(128)
WindowedGraphAttention_kernel(const float* __restrict__ x1,
                              const float* __restrict__ x2,
                              const float* __restrict__ wq, const float* __restrict__ bq,
                              const float* __restrict__ wk, const float* __restrict__ bk,
                              const float* __restrict__ wv, const float* __restrict__ bv,
                              const float* __restrict__ wp, const float* __restrict__ bp,
                              const float* __restrict__ gamma,
                              float* __restrict__ out) {
  extern __shared__ char smem[];
  float*    sX1 = (float*)   (smem + OFF_SX1);   // channel-major x1 tile
  uint16_t* sWq = (uint16_t*)(smem + OFF_SWQ);
  uint16_t* sWk = (uint16_t*)(smem + OFF_SWK);
  uint16_t* sWv = (uint16_t*)(smem + OFF_SWV);
  uint16_t* sVt = (uint16_t*)(smem + OFF_SVT);
  uint16_t* sP  = (uint16_t*)(smem + OFF_SP);
  float*    sX2 = (float*)   (smem + OFF_SX2);   // channel-major x2 tile
  uint16_t* sK  = (uint16_t*)(smem + OFF_SK);
  uint16_t* sQ  = (uint16_t*)(smem + OFF_SQ);
  float*    sO  = (float*)   (smem + OFF_SO);

  const int tid  = threadIdx.x;
  const int lane = tid & 31;
  const int wave = tid >> 5;
  const int col  = lane & 15;            // N-column within a 16-wide tile
  const int rb   = (lane >> 4) << 3;     // row base within D-tile: 0 or 8
  const int tok0 = wave * 16;            // this wave's 16 tokens

  // window decode: 1024 windows per batch image (32 x 32 of 8x8)
  const int win = blockIdx.x;
  const int b   = win >> 10;
  const int wh  = (win >> 5) & 31;
  const int ww  = win & 31;
  const long HW = 65536;                 // 256*256
  const long gbase = ((long)b * 128) * HW + (long)wh * 2048 + (long)ww * 8;

  // ---- kick off TDM DMAs for both input tiles (wave 0, EXEC-independent) ----
  if (wave == 0) {
    tdm_load_tile((uint32_t)(uintptr_t)sX1, x1 + gbase);
    tdm_load_tile((uint32_t)(uintptr_t)sX2, x2 + gbase);
  }

  // ---- stage weights (fp32 -> bf16) into LDS, overlapping the DMA ----------
  for (int i = tid; i < 2048; i += 128) {
    sWq[i] = (uint16_t)f2bf1(wq[i]);
    sWk[i] = (uint16_t)f2bf1(wk[i]);
  }
  for (int i = tid; i < 16384; i += 128) sWv[i] = (uint16_t)f2bf1(wv[i]);

  if (wave == 0) __builtin_amdgcn_s_wait_tensorcnt(0);
  __syncthreads();

  // ---- projections: q = X1 Wq^T + bq ; k = X2 Wk^T + bk ; v = X2 Wv^T + bv --
  Acc qacc, kacc, vacc[8];
  {
    const float bqv = bq[col];
    const float bkv = bk[col];
#pragma unroll
    for (int g = 0; g < 8; ++g) { qacc.f[g] = bqv; kacc.f[g] = bkv; }
#pragma unroll
    for (int nt = 0; nt < 8; ++nt) {
      const float bvv = bv[nt * 16 + col];
#pragma unroll
      for (int g = 0; g < 8; ++g) vacc[nt].f[g] = bvv;
    }
#pragma unroll
    for (int kc = 0; kc < 4; ++kc) {
      v16bf a1f = load_a_cm(sX1, lane, tok0, kc * 32);
      qacc.v = wmma_bf16(a1f, load_bf16_frag(sWq, 128, lane, kc * 32), qacc.v);
      v16bf a2f = load_a_cm(sX2, lane, tok0, kc * 32);
      kacc.v = wmma_bf16(a2f, load_bf16_frag(sWk, 128, lane, kc * 32), kacc.v);
#pragma unroll
      for (int nt = 0; nt < 8; ++nt)
        vacc[nt].v = wmma_bf16(a2f, load_bf16_frag(sWv + nt * 2048, 128, lane, kc * 32),
                               vacc[nt].v);
    }
  }
  // store q (per-wave 16x16), k (64x16) row-major; v transposed sVt[c][t]
  {
    uint16_t* qd = sQ + wave * 256;
#pragma unroll
    for (int g = 0; g < 8; ++g) {
      qd[(rb + g) * 16 + col]         = (uint16_t)f2bf1(qacc.f[g]);
      sK[(tok0 + rb + g) * 16 + col]  = (uint16_t)f2bf1(kacc.f[g]);
    }
#pragma unroll
    for (int nt = 0; nt < 8; ++nt) {
      v4u pk;
      pk.x = pack_bf(vacc[nt].f[0], vacc[nt].f[1]);
      pk.y = pack_bf(vacc[nt].f[2], vacc[nt].f[3]);
      pk.z = pack_bf(vacc[nt].f[4], vacc[nt].f[5]);
      pk.w = pack_bf(vacc[nt].f[6], vacc[nt].f[7]);
      *(v4u*)(sVt + (nt * 16 + col) * SVT_STRIDE + tok0 + rb) = pk;
    }
  }
  __syncthreads();

  // ---- attn = q k^T / sqrt(Cq), softmax over 64 columns ---------------------
  Acc att[4];
  {
    v16bf aq = load_bf16_frag16(sQ + wave * 256, 16, lane);
    v8f z = {};
#pragma unroll
    for (int nt = 0; nt < 4; ++nt)
      att[nt].v = wmma_bf16(aq, load_bf16_frag16(sK + nt * 256, 16, lane), z);

    float mrow[8], ssum[8];
#pragma unroll
    for (int g = 0; g < 8; ++g) {
      float m = att[0].f[g] * 0.25f;
#pragma unroll
      for (int nt = 0; nt < 4; ++nt) {
        att[nt].f[g] *= 0.25f;
        m = fmaxf(m, att[nt].f[g]);
      }
      mrow[g] = m;
    }
#pragma unroll
    for (int s = 1; s <= 8; s <<= 1)
#pragma unroll
      for (int g = 0; g < 8; ++g)
        mrow[g] = fmaxf(mrow[g], __shfl_xor(mrow[g], s, 32));
#pragma unroll
    for (int g = 0; g < 8; ++g) {
      float acc = 0.f;
#pragma unroll
      for (int nt = 0; nt < 4; ++nt) {
        float e = __expf(att[nt].f[g] - mrow[g]);
        att[nt].f[g] = e;
        acc += e;
      }
      ssum[g] = acc;
    }
#pragma unroll
    for (int s = 1; s <= 8; s <<= 1)
#pragma unroll
      for (int g = 0; g < 8; ++g)
        ssum[g] += __shfl_xor(ssum[g], s, 32);

    uint16_t* pd = sP + wave * 1024;
#pragma unroll
    for (int g = 0; g < 8; ++g) {
      const float rinv = 1.0f / ssum[g];
#pragma unroll
      for (int nt = 0; nt < 4; ++nt)
        pd[(rb + g) * 64 + nt * 16 + col] = (uint16_t)f2bf1(att[nt].f[g] * rinv);
    }
  }
  __syncthreads();   // also guarantees sX2/sK are dead before sO aliasing

  // ---- o = P (16x64) x V (64x128) ------------------------------------------
  Acc occ[8];
#pragma unroll
  for (int nt = 0; nt < 8; ++nt) { v8f z = {}; occ[nt].v = z; }
#pragma unroll
  for (int kc = 0; kc < 2; ++kc) {
    v16bf ap = load_bf16_frag(sP + wave * 1024, 64, lane, kc * 32);
#pragma unroll
    for (int nt = 0; nt < 8; ++nt)
      occ[nt].v = wmma_bf16(ap, load_bf16_frag(sVt + nt * 16 * SVT_STRIDE,
                                               SVT_STRIDE, lane, kc * 32),
                            occ[nt].v);
  }
  // stage o to LDS channel-major (stride 65 => conflict-free epilogue reads)
#pragma unroll
  for (int nt = 0; nt < 8; ++nt) {
    float* d = sO + (nt * 16 + col) * SO_STRIDE + tok0 + rb;
#pragma unroll
    for (int g = 0; g < 8; ++g) d[g] = occ[nt].f[g];
  }
  __syncthreads();

  // ---- epilogue: out = x1 + gamma * (o * wp + bp), coalesced 32B stores -----
  {
    const float g0 = gamma[0];
    for (int r = tid; r < 1024; r += 128) {
      const int c = r >> 3, h = r & 7;
      const float wpc = wp[c], bpc = bp[c];
      const float* po = sO  + c * SO_STRIDE + h * 8;
      const float* px = sX1 + c * SXC_STRIDE + h * 8;   // channel-major residual
      float vbuf[8];
#pragma unroll
      for (int w = 0; w < 8; ++w)
        vbuf[w] = px[w] + g0 * (po[w] * wpc + bpc);
      const long ga = gbase + (long)c * HW + (long)h * 256;
      float4 o0 = { vbuf[0], vbuf[1], vbuf[2], vbuf[3] };
      float4 o1 = { vbuf[4], vbuf[5], vbuf[6], vbuf[7] };
      *(float4*)(out + ga)     = o0;
      *(float4*)(out + ga + 4) = o1;
    }
  }
}

extern "C" void kernel_launch(void* const* d_in, const int* in_sizes, int n_in,
                              void* d_out, int out_size, void* d_ws, size_t ws_size,
                              hipStream_t stream) {
  (void)in_sizes; (void)n_in; (void)out_size; (void)d_ws; (void)ws_size;
  const float* x1    = (const float*)d_in[0];
  const float* x2    = (const float*)d_in[1];
  const float* wq    = (const float*)d_in[2];
  const float* bq    = (const float*)d_in[3];
  const float* wk    = (const float*)d_in[4];
  const float* bk    = (const float*)d_in[5];
  const float* wv    = (const float*)d_in[6];
  const float* bv    = (const float*)d_in[7];
  const float* wp    = (const float*)d_in[8];
  const float* bp    = (const float*)d_in[9];
  const float* gamma = (const float*)d_in[10];
  float* out = (float*)d_out;

  hipFuncSetAttribute((const void*)WindowedGraphAttention_kernel,
                      hipFuncAttributeMaxDynamicSharedMemorySize, SMEM_BYTES);

  dim3 grid(4096);   // 4 batches * 32 * 32 windows
  dim3 block(128);   // 4 wave32s, one 16-token M-tile each
  hipLaunchKernelGGL(WindowedGraphAttention_kernel, grid, block, SMEM_BYTES, stream,
                     x1, x2, wq, bq, wk, bk, wv, bv, wp, bp, gamma, out);
}
